// ContextAwareEncoder_44255343018528
// MI455X (gfx1250) — compile-verified, matrix operand order checked by reading, not run on hardware
//
#include <hip/hip_runtime.h>
#include <hip/hip_bf16.h>

// ---------------------------------------------------------------------------
// CDNA5 (gfx1250) bf16 WMMA implementation of ContextAwareEncoder.
// GEMMs use v_wmma_f32_16x16x32_bf16 (wave32, 8 waves / 256-thread block) with
// 64x128 block tiles (4 WMMA per wave per K-step) and async global->LDS
// copies (ASYNCcnt) for the non-transposing tile feeds.
// Attention (scores -> softmax -> ctx) is fused per 16-row block in LDS.
// ---------------------------------------------------------------------------

typedef __bf16 bhalf;
typedef __attribute__((ext_vector_type(8)))  __bf16 v8bh;
typedef __attribute__((ext_vector_type(16))) __bf16 v16bh;
typedef __attribute__((ext_vector_type(8)))  float  v8f;

__device__ __forceinline__ v8f zero8() {
  v8f z = {0.f, 0.f, 0.f, 0.f, 0.f, 0.f, 0.f, 0.f};
  return z;
}

__device__ __forceinline__ v8f wmma_bf16(v16bh a, v16bh b, v8f c) {
  // (neg_a, A, neg_b, B, c_mod, C, reuse_a, reuse_b)
  return __builtin_amdgcn_wmma_f32_16x16x32_bf16(false, a, false, b, (short)0, c,
                                                 false, false);
}

// Async 16B global->LDS copy (GLOBAL_LOAD_ASYNC_TO_LDS_B128, ASYNCcnt).
// lds_addr: wave-relative LDS byte address (= low 32 bits of a generic
// __shared__ pointer per ISA flat addressing).
__device__ __forceinline__ void async_copy_b128(unsigned lds_addr,
                                                const void* gptr) {
  asm volatile("global_load_async_to_lds_b128 %0, %1, off"
               :: "v"(lds_addr), "v"(gptr)
               : "memory");
}
__device__ __forceinline__ void wait_async() {
  asm volatile("s_wait_asynccnt 0x0" ::: "memory");
}

// Load one lane's 16-element A/B fragment from an LDS tile stored row-major
// with K contiguous. ISA 16-bit layout: elems 0..7 -> K = k0 + 8*laneHi + e,
// elems 8..15 -> K = k0 + 16 + 8*laneHi + e. Two 16B ds loads per lane.
__device__ __forceinline__ v16bh ld_frag(const bhalf* base, int row_stride,
                                         int row, int k0, int laneHi) {
  const bhalf* p = base + row * row_stride + k0 + 8 * laneHi;
  v8bh lo = *(const v8bh*)p;
  v8bh hi = *(const v8bh*)(p + 16);
  return __builtin_shufflevector(lo, hi, 0, 1, 2, 3, 4, 5, 6, 7, 8, 9, 10, 11,
                                 12, 13, 14, 15);
}

// ---------------------------------------------------------------------------
// Tiled GEMM: C[b] = A (M x K, shared) * B[b] (K x N), fp32 out.
// Block tile 64x128, K-tile 32; 8 waves, 4 16x16 output tiles per wave.
// A tile is copied via async global->LDS; B tile is transposed through VGPRs.
// ---------------------------------------------------------------------------
__global__ __launch_bounds__(256) void gemm_wmma(
    const bhalf* __restrict__ A, const bhalf* __restrict__ B,
    float* __restrict__ C, int M, int N, int K, int strideB, int strideC) {
  __shared__ bhalf sA[64 * 40];   // 64 rows (M), 32 K + pad 8
  __shared__ bhalf sB[128 * 40];  // 128 rows (N, transposed), 32 K + pad 8
  const int t = threadIdx.x;
  const int lane = t & 31, w = t >> 5;
  const int laneHi = (lane >> 4) & 1, rsel = lane & 15;
  const int m0 = blockIdx.y * 64, n0 = blockIdx.x * 128;
  const bhalf* Bb = B + (size_t)blockIdx.z * strideB;
  float* Cb = C + (size_t)blockIdx.z * strideC;
  const int tm = w >> 1, tnb = (w & 1) * 4;

  const int ra = t >> 2, ca = (t & 3) * 8;     // A tile loader mapping
  const int rb = t >> 3, cbB = (t & 7) * 16;   // B tile loader mapping
  const unsigned ldsA = (unsigned)(uintptr_t)(void*)&sA[ra * 40 + ca];

  v8f acc[4] = {zero8(), zero8(), zero8(), zero8()};
  const int ksteps = K >> 5;
  for (int kk = 0; kk < ksteps; ++kk) {
    const int k0 = kk << 5;
    // A (row-major M x K): async DMA straight into LDS, no transpose needed
    async_copy_b128(ldsA, &A[(size_t)(m0 + ra) * K + k0 + ca]);
    // B (row-major K x N): load N-contiguous, store transposed (N-major)
    v8bh bv0 = *(const v8bh*)&Bb[(size_t)(k0 + rb) * N + n0 + cbB];
    v8bh bv1 = *(const v8bh*)&Bb[(size_t)(k0 + rb) * N + n0 + cbB + 8];
#pragma unroll
    for (int j = 0; j < 8; ++j) sB[(cbB + j) * 40 + rb] = bv0[j];
#pragma unroll
    for (int j = 0; j < 8; ++j) sB[(cbB + 8 + j) * 40 + rb] = bv1[j];
    if (kk + 1 < ksteps)  // global_prefetch_b8 for next B tile
      __builtin_prefetch(&Bb[(size_t)(k0 + 32 + rb) * N + n0 + cbB], 0, 3);
    wait_async();
    __syncthreads();

    v16bh af = ld_frag(sA, 40, tm * 16 + rsel, 0, laneHi);
#pragma unroll
    for (int j = 0; j < 4; ++j) {
      v16bh bf = ld_frag(sB, 40, (tnb + j) * 16 + rsel, 0, laneHi);
      acc[j] = wmma_bf16(af, bf, acc[j]);
    }
    __syncthreads();
  }
  // D layout: elem e -> row e + 8*laneHi, col = lane&15
#pragma unroll
  for (int j = 0; j < 4; ++j) {
#pragma unroll
    for (int e = 0; e < 8; ++e) {
      const int row = m0 + tm * 16 + e + 8 * laneHi;
      Cb[(size_t)row * N + n0 + (tnb + j) * 16 + rsel] = acc[j][e];
    }
  }
}

// ---------------------------------------------------------------------------
// Fused contextual descriptor. f = cat[b, 0:256, :] (256 x 1024 bf16).
// Per block: 16 query rows n0..n0+15.
//   S(16x1024) = f^T[rows] * f   (WMMA, staged in LDS fp32)
//   softmax rows in LDS -> bf16 probs
//   O(16x256)  = P * f^T         (WMMA, V tiles fed by async global->LDS)
// O written raw (n-major, c contiguous) into cat[b, 256:512, :] (the
// reference's .view reinterpret is a flat byte reinterpret).
// ---------------------------------------------------------------------------
#define ATTN_SMEM 194816
__global__ __launch_bounds__(256) void attn_wmma(bhalf* cat) {
  extern __shared__ char smem[];
  bhalf* sA = (bhalf*)smem;              // 16  x 264  (f^T query rows, K=c)
  bhalf* sB = sA + 16 * 264;             // 128 x 264  (f^T key rows,   K=c)
  float* sS = (float*)(sB + 128 * 264);  // 16  x 1024 (scores)
  bhalf* sP = (bhalf*)(sS + 16 * 1024);  // 16  x 1024 (probs)
  bhalf* sV = sP + 16 * 1024;            // 256 x 40   (V tile, K=m)

  const int t = threadIdx.x;
  const int lane = t & 31, w = t >> 5;
  const int laneHi = (lane >> 4) & 1, rsel = lane & 15;
  const int n0 = blockIdx.x * 16;
  bhalf* f = cat + (size_t)blockIdx.y * (512 * 1024);
  bhalf* gs = f + 256 * 1024;

  {  // stage A = f^T rows n0..n0+15 over all 256 channels (thread t = channel)
    const bhalf* fr = f + (size_t)t * 1024 + n0;
#pragma unroll
    for (int j = 0; j < 16; ++j) sA[j * 264 + t] = fr[j];
  }
  __syncthreads();

  // ---- phase 1: scores, 8 column blocks of 128 keys, 1 16-col tile/wave ----
  for (int mo = 0; mo < 8; ++mo) {
    const int m0 = mo * 128;
    const bhalf* fr = f + (size_t)t * 1024 + m0;
#pragma unroll
    for (int j8 = 0; j8 < 16; ++j8) {
      v8bh v = *(const v8bh*)(fr + j8 * 8);
#pragma unroll
      for (int e = 0; e < 8; ++e) sB[(j8 * 8 + e) * 264 + t] = v[e];
    }
    __syncthreads();
    v8f acc = zero8();
    const int colb = w * 16;
#pragma unroll
    for (int kb = 0; kb < 8; ++kb) {  // K = 256 channels
      v16bh af = ld_frag(sA, 264, rsel, kb * 32, laneHi);
      v16bh bf = ld_frag(sB, 264, colb + rsel, kb * 32, laneHi);
      acc = wmma_bf16(af, bf, acc);
    }
#pragma unroll
    for (int j = 0; j < 8; ++j)
      sS[(j + 8 * laneHi) * 1024 + m0 + colb + rsel] = acc[j];
    __syncthreads();
  }

  // ---- phase 2: row softmax (2 rows per wave), pack bf16 probs ----
  for (int rr = w * 2; rr < w * 2 + 2; ++rr) {
    float* row = sS + rr * 1024;
    float mx = -3.4e38f;
    for (int i = lane; i < 1024; i += 32) mx = fmaxf(mx, row[i]);
#pragma unroll
    for (int off = 16; off; off >>= 1) mx = fmaxf(mx, __shfl_xor(mx, off, 32));
    float s = 0.f;
    for (int i = lane; i < 1024; i += 32) {
      float e = __expf(row[i] - mx);
      row[i] = e;
      s += e;
    }
#pragma unroll
    for (int off = 16; off; off >>= 1) s += __shfl_xor(s, off, 32);
    const float inv = 1.f / s;
    bhalf* prow = sP + rr * 1024;
    for (int i = lane; i < 1024; i += 32) prow[i] = (bhalf)(row[i] * inv);
  }
  __syncthreads();

  // ---- phase 3: O = P * V, stream 32-wide K(m) tiles of V = f^T ----
  v8f o0 = zero8(), o1 = zero8();
  for (int kt = 0; kt < 32; ++kt) {
    const bhalf* fr = f + (size_t)t * 1024 + kt * 32;
#pragma unroll
    for (int j8 = 0; j8 < 4; ++j8)  // V tile row c = t, K(m) contiguous: async
      async_copy_b128((unsigned)(uintptr_t)(void*)&sV[t * 40 + j8 * 8],
                      fr + j8 * 8);
    wait_async();
    __syncthreads();
    v16bh af = ld_frag(sP, 1024, rsel, kt * 32, laneHi);
    v16bh b0 = ld_frag(sV, 40, 32 * w + rsel, 0, laneHi);
    v16bh b1 = ld_frag(sV, 40, 32 * w + 16 + rsel, 0, laneHi);
    o0 = wmma_bf16(af, b0, o0);
    o1 = wmma_bf16(af, b1, o1);
    __syncthreads();
  }
#pragma unroll
  for (int j = 0; j < 8; ++j) {
    const int n = n0 + j + 8 * laneHi;
    gs[(size_t)n * 256 + 32 * w + rsel] = (bhalf)o0[j];
    gs[(size_t)n * 256 + 32 * w + 16 + rsel] = (bhalf)o1[j];
  }
}

// ---------------------------------------------------------------------------
// Elementwise / BN helper kernels (deterministic fixed-order reductions).
// ---------------------------------------------------------------------------
__global__ void cvt_bf16(const float* __restrict__ s, bhalf* __restrict__ d,
                         int n) {
  const int i = blockIdx.x * 256 + threadIdx.x;
  if (i < n) d[i] = (bhalf)s[i];
}

// One block per (channel, batch): reduce 1024 spatial values.
__global__ __launch_bounds__(256) void bn_stats(const float* __restrict__ src,
                                                float* __restrict__ part,
                                                int C) {
  __shared__ float rs[256], rq[256];
  const int c = blockIdx.x, b = blockIdx.y, t = threadIdx.x;
  const float* p = src + ((size_t)b * C + c) * 1024;
  float s = 0.f, q = 0.f;
  for (int i = t; i < 1024; i += 256) {
    float v = p[i];
    s += v;
    q += v * v;
  }
  rs[t] = s;
  rq[t] = q;
  __syncthreads();
  for (int off = 128; off; off >>= 1) {
    if (t < off) {
      rs[t] += rs[t + off];
      rq[t] += rq[t + off];
    }
    __syncthreads();
  }
  if (t == 0) {
    part[(c * 16 + b) * 2] = rs[0];
    part[(c * 16 + b) * 2 + 1] = rq[0];
  }
}

__global__ void bn_finalize(const float* __restrict__ part,
                            const float* __restrict__ gamma,
                            const float* __restrict__ beta,
                            float* __restrict__ scale, float* __restrict__ bias,
                            int C) {
  const int c = blockIdx.x * 256 + threadIdx.x;
  if (c >= C) return;
  float s = 0.f, q = 0.f;
  for (int b = 0; b < 16; ++b) {
    s += part[(c * 16 + b) * 2];
    q += part[(c * 16 + b) * 2 + 1];
  }
  const float invn = 1.0f / 16384.0f;  // b*h*w = 16*32*32 (biased var)
  const float mean = s * invn;
  const float var = fmaxf(q * invn - mean * mean, 0.f);
  const float sc = gamma[c] * rsqrtf(var + 1e-5f);
  scale[c] = sc;
  bias[c] = beta[c] - mean * sc;
}

__global__ void bn_apply_bf16(const float* __restrict__ src,
                              bhalf* __restrict__ dst,
                              const float* __restrict__ scale,
                              const float* __restrict__ bias, int C,
                              int dstBatchStride, int chanOff, int relu,
                              int total) {
  const int i = blockIdx.x * 256 + threadIdx.x;
  if (i >= total) return;
  const int cn = C << 10;
  const int b = i / cn, rem = i - b * cn, c = rem >> 10, n = rem & 1023;
  float v = src[i] * scale[c] + bias[c];
  if (relu) v = fmaxf(v, 0.f);
  dst[(size_t)b * dstBatchStride + ((size_t)(c + chanOff) << 10) + n] = (bhalf)v;
}

__global__ void bn_apply_f32(float* __restrict__ data,
                             const float* __restrict__ scale,
                             const float* __restrict__ bias, int C, int total) {
  const int i = blockIdx.x * 256 + threadIdx.x;
  if (i >= total) return;
  const int c = (i >> 10) % C;
  data[i] = data[i] * scale[c] + bias[c];
}

// ---------------------------------------------------------------------------
extern "C" void kernel_launch(void* const* d_in, const int* in_sizes, int n_in,
                              void* d_out, int out_size, void* d_ws,
                              size_t ws_size, hipStream_t stream) {
  (void)in_sizes; (void)n_in; (void)out_size; (void)ws_size;
  const float* x     = (const float*)d_in[0];
  const float* w_in  = (const float*)d_in[1];
  const float* g1    = (const float*)d_in[2];
  const float* b1    = (const float*)d_in[3];
  const float* w_emb = (const float*)d_in[4];
  const float* g2    = (const float*)d_in[5];
  const float* b2    = (const float*)d_in[6];
  const float* w_out = (const float*)d_in[7];
  const float* g3    = (const float*)d_in[8];
  const float* b3    = (const float*)d_in[9];
  float* out = (float*)d_out;

  const int B = 16, C0 = 640, C1 = 256, C2 = 512, C3 = 1024, N = 1024;

  char* ws = (char*)d_ws;
  size_t off = 0;
  auto alloc = [&](size_t bytes) {
    char* p = ws + off;
    off += (bytes + 255) & ~(size_t)255;
    return p;
  };
  bhalf* xb   = (bhalf*)alloc((size_t)B * C0 * N * 2);        // bf16 input
  bhalf* w1b  = (bhalf*)alloc((size_t)C1 * C0 * 2);
  bhalf* w2b  = (bhalf*)alloc((size_t)C2 * 2 * C1 * 2);
  bhalf* w3b  = (bhalf*)alloc((size_t)C3 * C2 * 2);
  float* h1p  = (float*)alloc((size_t)B * C1 * N * 4);        // pre-BN conv1
  bhalf* cat  = (bhalf*)alloc((size_t)B * 2 * C1 * N * 2);    // [h1 ; gs] bf16
  float* h2p  = (float*)alloc((size_t)B * C2 * N * 4);        // pre-BN conv2
  bhalf* h2b  = (bhalf*)alloc((size_t)B * C2 * N * 2);        // h2 bf16
  float* part = (float*)alloc((size_t)C3 * 16 * 2 * 4);       // BN partials
  float* scl  = (float*)alloc((size_t)C3 * 4);
  float* bia  = (float*)alloc((size_t)C3 * 4);

  // --- precision downconvert (bf16 operands for WMMA) ---
  cvt_bf16<<<(B * C0 * N + 255) / 256, 256, 0, stream>>>(x, xb, B * C0 * N);
  cvt_bf16<<<(C1 * C0 + 255) / 256, 256, 0, stream>>>(w_in, w1b, C1 * C0);
  cvt_bf16<<<(C2 * 2 * C1 + 255) / 256, 256, 0, stream>>>(w_emb, w2b, C2 * 2 * C1);
  cvt_bf16<<<(C3 * C2 + 255) / 256, 256, 0, stream>>>(w_out, w3b, C3 * C2);

  // --- stage 1: conv1 + BN + ReLU -> cat[:, 0:256] (bf16) ---
  gemm_wmma<<<dim3(N / 128, C1 / 64, B), 256, 0, stream>>>(
      w1b, xb, h1p, C1, N, C0, C0 * N, C1 * N);
  bn_stats<<<dim3(C1, B), 256, 0, stream>>>(h1p, part, C1);
  bn_finalize<<<(C1 + 255) / 256, 256, 0, stream>>>(part, g1, b1, scl, bia, C1);
  bn_apply_bf16<<<(B * C1 * N + 255) / 256, 256, 0, stream>>>(
      h1p, cat, scl, bia, C1, 2 * C1 * N, 0, 1, B * C1 * N);

  // --- stage 2: fused contextual descriptor -> cat[:, 256:512] ---
  attn_wmma<<<dim3(N / 16, B), 256, ATTN_SMEM, stream>>>(cat);

  // --- stage 3: conv2 + BN + ReLU ---
  gemm_wmma<<<dim3(N / 128, C2 / 64, B), 256, 0, stream>>>(
      w2b, cat, h2p, C2, N, 2 * C1, 2 * C1 * N, C2 * N);
  bn_stats<<<dim3(C2, B), 256, 0, stream>>>(h2p, part, C2);
  bn_finalize<<<(C2 + 255) / 256, 256, 0, stream>>>(part, g2, b2, scl, bia, C2);
  bn_apply_bf16<<<(B * C2 * N + 255) / 256, 256, 0, stream>>>(
      h2p, h2b, scl, bia, C2, C2 * N, 0, 1, B * C2 * N);

  // --- stage 4: conv3 -> d_out, then BN in place ---
  gemm_wmma<<<dim3(N / 128, C3 / 64, B), 256, 0, stream>>>(
      w3b, h2b, out, C3, N, C2, C2 * N, C3 * N);
  bn_stats<<<dim3(C3, B), 256, 0, stream>>>(out, part, C3);
  bn_finalize<<<(C3 + 255) / 256, 256, 0, stream>>>(part, g3, b3, scl, bia, C3);
  bn_apply_f32<<<(B * C3 * N + 255) / 256, 256, 0, stream>>>(out, scl, bia, C3,
                                                             B * C3 * N);
}
